// TransformerBlock_80487687127092
// MI455X (gfx1250) — matrix operand order repaired and checked by reading, not compiled, not run on hardware
//
#include <hip/hip_runtime.h>
#include <math.h>

#define DEV __device__ __forceinline__

typedef __attribute__((ext_vector_type(16))) __bf16 bfx16;
typedef __attribute__((ext_vector_type(8)))  float  fx8;

// ---------- bf16 bit helpers (manual RNE convert; avoids relying on __bf16 arith) ----------
union BfBits { unsigned short u; __bf16 h; };

DEV unsigned short f2bfu(float f) {
  union { float f; unsigned u; } a; a.f = f;
  unsigned r = a.u + 0x7FFFu + ((a.u >> 16) & 1u);
  return (unsigned short)(r >> 16);
}
DEV __bf16 f2bf(float f) { BfBits b; b.u = f2bfu(f); return b.h; }

// ---------- WMMA fragment loaders (layouts per CDNA5 ISA 7.12.2) ----------
union FragU { uint4 q[2]; bfx16 v; unsigned short s[16]; };

// A: 16x32 bf16, row-major source A[m][k], lda in elements.
// lane<16: M=lane, K = k0+0..7 (elems 0..7) and k0+16..23 (elems 8..15)
// lane>=16: M=lane-16, K = k0+8..15 and k0+24..31
DEV bfx16 load_a_frag(const __bf16* A, int lda, int row0, int k0, int lane) {
  int m  = lane & 15;
  int kb = k0 + ((lane >> 4) << 3);
  const __bf16* p = A + (size_t)(row0 + m) * lda + kb;
  FragU u;
  u.q[0] = *reinterpret_cast<const uint4*>(p);
  u.q[1] = *reinterpret_cast<const uint4*>(p + 16);
  return u.v;
}

// B: 32x16 bf16 logical B[k][n]; source stored transposed Bt[n][k] (ldb elems).
// lane<16: N=lane, K = k0+0..7 and k0+16..23 ; lane>=16: N=lane-16, K = k0+8..15 and k0+24..31
DEV bfx16 load_b_frag(const __bf16* Bt, int ldb, int col0, int k0, int lane) {
  int n  = lane & 15;
  int kb = k0 + ((lane >> 4) << 3);
  const __bf16* p = Bt + (size_t)(col0 + n) * ldb + kb;
  FragU u;
  u.q[0] = *reinterpret_cast<const uint4*>(p);
  u.q[1] = *reinterpret_cast<const uint4*>(p + 16);
  return u.v;
}

DEV fx8 wmma_bf16(bfx16 a, bfx16 b, fx8 c) {
  return __builtin_amdgcn_wmma_f32_16x16x32_bf16(false, a, false, b, (short)0, c, false, false);
}

// A-fragment gather from LDS (unsigned short array, pitch in elements)
DEV bfx16 lds_a_frag(const unsigned short* P, int pitch, int k0, int lane) {
  int m  = lane & 15;
  int kb = k0 + ((lane >> 4) << 3);
  FragU u;
#pragma unroll
  for (int e = 0; e < 8; ++e) {
    u.s[e]     = P[m * pitch + kb + e];
    u.s[e + 8] = P[m * pitch + kb + 16 + e];
  }
  return u.v;
}

// ============================ LayerNorm (fp32 in -> bf16 out) ============================
__global__ __launch_bounds__(256) void ln_bf16_kernel(const float* __restrict__ x,
                                                      const float* __restrict__ g,
                                                      const float* __restrict__ be,
                                                      __bf16* __restrict__ out) {
  const int row = blockIdx.x, tid = threadIdx.x;
  const float4 v = reinterpret_cast<const float4*>(x + (size_t)row * 1024)[tid];
  float s = v.x + v.y + v.z + v.w;
  float q = v.x * v.x + v.y * v.y + v.z * v.z + v.w * v.w;
#pragma unroll
  for (int m = 1; m < 32; m <<= 1) { s += __shfl_xor(s, m, 32); q += __shfl_xor(q, m, 32); }
  __shared__ float ssum[8], ssq[8];
  const int wave = tid >> 5, lane = tid & 31;
  if (lane == 0) { ssum[wave] = s; ssq[wave] = q; }
  __syncthreads();
  if (tid == 0) {
    float ts = 0.f, tq = 0.f;
#pragma unroll
    for (int w = 0; w < 8; ++w) { ts += ssum[w]; tq += ssq[w]; }
    const float mu = ts * (1.0f / 1024.0f);
    const float var = tq * (1.0f / 1024.0f) - mu * mu;
    ssum[0] = mu; ssq[0] = rsqrtf(var + 1e-5f);
  }
  __syncthreads();
  const float mu = ssum[0], rs = ssq[0];
  const float4 gv = reinterpret_cast<const float4*>(g)[tid];
  const float4 bv = reinterpret_cast<const float4*>(be)[tid];
  __bf16* op = out + (size_t)row * 1024 + tid * 4;
  op[0] = f2bf((v.x - mu) * rs * gv.x + bv.x);
  op[1] = f2bf((v.y - mu) * rs * gv.y + bv.y);
  op[2] = f2bf((v.z - mu) * rs * gv.z + bv.z);
  op[3] = f2bf((v.w - mu) * rs * gv.w + bv.w);
}

// ===================== Weight convert: fp32 [K,N] -> bf16 transposed [N,K] ================
__global__ void wT_kernel(const float* __restrict__ src, __bf16* __restrict__ dst,
                          int K_, int N_) {
  const size_t total = (size_t)K_ * (size_t)N_;
  for (size_t i = (size_t)blockIdx.x * blockDim.x + threadIdx.x; i < total;
       i += (size_t)gridDim.x * blockDim.x) {
    const size_t n = i / (size_t)K_, k = i % (size_t)K_;
    dst[i] = f2bf(src[k * (size_t)N_ + n]);
  }
}

// per-head QKV weights: src[h, d, e] (16,1024,64) -> dst[(h*64+e)*1024 + d]
__global__ void qkvT_kernel(const float* __restrict__ src, __bf16* __restrict__ dst) {
  const size_t total = 1024u * 1024u;
  for (size_t i = (size_t)blockIdx.x * blockDim.x + threadIdx.x; i < total;
       i += (size_t)gridDim.x * blockDim.x) {
    const int n = (int)(i >> 10), k = (int)(i & 1023);
    const int h = n >> 6, e = n & 63;
    dst[i] = f2bf(src[((size_t)h << 16) + ((size_t)k << 6) + e]);
  }
}

// =============================== Tiled WMMA GEMM ==========================================
// C[M,N] = A[M,K](bf16,row) @ Bt[N,K](bf16)  (+bias, relu, +residual); 128x128 block,
// 8 waves (4x2), each wave 32x64 = 2x4 WMMA tiles, k-step 32.
template <bool BIAS, bool RELU, bool RES, bool SF32, bool SBF, bool SVT>
__global__ __launch_bounds__(256) void gemm_bf16_kernel(const __bf16* __restrict__ A,
                                                        const __bf16* __restrict__ Bt,
                                                        const float* __restrict__ bias,
                                                        const float* __restrict__ res,
                                                        float* __restrict__ outF,
                                                        __bf16* __restrict__ outB,
                                                        int M, int N, int K) {
  const int lane = threadIdx.x & 31, wave = threadIdx.x >> 5;
  const int rowBase = blockIdx.y * 128 + (wave >> 1) * 32;
  const int colBase = blockIdx.x * 128 + (wave & 1) * 64;
  (void)M;

  fx8 acc[2][4];
#pragma unroll
  for (int mt = 0; mt < 2; ++mt)
#pragma unroll
    for (int nt = 0; nt < 4; ++nt) acc[mt][nt] = (fx8)(0.0f);

  for (int k0 = 0; k0 < K; k0 += 32) {
    if (k0 + 32 < K) {  // hint next k-tile toward the caches (global_prefetch_b8)
      __builtin_prefetch(A + (size_t)(rowBase + lane) * K + k0 + 32, 0, 1);
      __builtin_prefetch(Bt + (size_t)(colBase + lane) * K + k0 + 32, 0, 1);
      __builtin_prefetch(Bt + (size_t)(colBase + 32 + lane) * K + k0 + 32, 0, 1);
    }
    bfx16 af[2], bfr[4];
#pragma unroll
    for (int mt = 0; mt < 2; ++mt) af[mt] = load_a_frag(A, K, rowBase + mt * 16, k0, lane);
#pragma unroll
    for (int nt = 0; nt < 4; ++nt) bfr[nt] = load_b_frag(Bt, K, colBase + nt * 16, k0, lane);
#pragma unroll
    for (int mt = 0; mt < 2; ++mt)
#pragma unroll
      for (int nt = 0; nt < 4; ++nt) acc[mt][nt] = wmma_bf16(af[mt], bfr[nt], acc[mt][nt]);
  }

#pragma unroll
  for (int mt = 0; mt < 2; ++mt)
#pragma unroll
    for (int nt = 0; nt < 4; ++nt)
#pragma unroll
      for (int i = 0; i < 8; ++i) {
        const int r = rowBase + mt * 16 + ((lane >> 4) << 3) + i;
        const int c = colBase + nt * 16 + (lane & 15);
        float v = acc[mt][nt][i];
        if (BIAS) v += bias[c];
        if (RELU) v = fmaxf(v, 0.0f);
        if (RES)  v += res[(size_t)r * N + c];
        if (SF32) outF[(size_t)r * N + c] = v;
        if (SBF) {
          if (SVT) {  // V stored per-head transposed: vt[((b*16+h)*64+e)*2048 + t]
            const int bb = r >> 11, tt = r & 2047, hh = c >> 6, ee = c & 63;
            outB[((((size_t)bb * 16 + hh) * 64) + ee) * 2048 + tt] = f2bf(v);
          } else {
            outB[(size_t)r * N + c] = f2bf(v);
          }
        }
      }
}

// =========================== Causal flash attention (WMMA) ================================
// grid: (T/128, B*H); block 256. Each wave owns 16 query rows; K/V blocks of 64.
__global__ __launch_bounds__(256) void attn_kernel(const __bf16* __restrict__ Q,
                                                   const __bf16* __restrict__ Kb,
                                                   const __bf16* __restrict__ Vt,
                                                   __bf16* __restrict__ Ob) {
  constexpr int T = 2048, Dm = 1024, HD = 64;
  constexpr int PITCH = 72;
  const int lane = threadIdx.x & 31, wave = threadIdx.x >> 5;
  const int bh = blockIdx.y, b = bh >> 4, h = bh & 15;
  const int qRow = blockIdx.x * 128 + wave * 16;

  const __bf16* Qh = Q + (size_t)b * T * Dm + h * HD;   // [T, HD], stride Dm
  const __bf16* Kh = Kb + (size_t)b * T * Dm + h * HD;  // [T, HD], stride Dm
  const __bf16* Vh = Vt + (size_t)bh * HD * T;          // [HD, T] (transposed), stride T

  __shared__ __align__(16) unsigned short Pl[8][16 * PITCH];
  unsigned short* Pw = &Pl[wave][0];

  bfx16 qf[2];
  qf[0] = load_a_frag(Qh, Dm, qRow, 0, lane);
  qf[1] = load_a_frag(Qh, Dm, qRow, 32, lane);

  fx8 o[4];
#pragma unroll
  for (int nt = 0; nt < 4; ++nt) o[nt] = (fx8)(0.0f);
  float mrun[8], lrun[8];
#pragma unroll
  for (int i = 0; i < 8; ++i) { mrun[i] = -INFINITY; lrun[i] = 0.0f; }

  const int sEnd = qRow + 15;
  for (int s0 = 0; s0 <= sEnd; s0 += 64) {
    fx8 S[4];
#pragma unroll
    for (int nt = 0; nt < 4; ++nt) S[nt] = (fx8)(0.0f);

    // S = (Q @ K^T) : k-dim = HD (2 steps of 32), n = key index
#pragma unroll
    for (int kk = 0; kk < 2; ++kk) {
      bfx16 kf[4];
#pragma unroll
      for (int nt = 0; nt < 4; ++nt) kf[nt] = load_b_frag(Kh, Dm, s0 + nt * 16, kk * 32, lane);
#pragma unroll
      for (int nt = 0; nt < 4; ++nt) S[nt] = wmma_bf16(qf[kk], kf[nt], S[nt]);
    }

    const bool diag = (s0 + 63 > qRow);
#pragma unroll
    for (int nt = 0; nt < 4; ++nt)
#pragma unroll
      for (int i = 0; i < 8; ++i) {
        float v = S[nt][i] * 0.03125f;  // D^-0.5 = 1/32 (scales by embed dim per reference)
        if (diag) {
          const int sg = s0 + nt * 16 + (lane & 15);
          const int tg = qRow + ((lane >> 4) << 3) + i;
          if (sg > tg) v = -INFINITY;
        }
        S[nt][i] = v;
      }

    // online softmax, write P (bf16) into this wave's LDS region
#pragma unroll
    for (int i = 0; i < 8; ++i) {
      float r = S[0][i];
#pragma unroll
      for (int nt = 1; nt < 4; ++nt) r = fmaxf(r, S[nt][i]);
#pragma unroll
      for (int m = 1; m < 16; m <<= 1) r = fmaxf(r, __shfl_xor(r, m, 32));
      const float mn = fmaxf(mrun[i], r);
      const float corr = __expf(mrun[i] - mn);
      lrun[i] *= corr;
      mrun[i] = mn;
#pragma unroll
      for (int nt = 0; nt < 4; ++nt) o[nt][i] *= corr;
      const int prow = ((lane >> 4) << 3) + i;
      float ps = 0.0f;
#pragma unroll
      for (int nt = 0; nt < 4; ++nt) {
        const float p = __expf(S[nt][i] - mn);
        ps += p;
        Pw[prow * PITCH + nt * 16 + (lane & 15)] = f2bfu(p);
      }
#pragma unroll
      for (int m = 1; m < 16; m <<= 1) ps += __shfl_xor(ps, m, 32);
      lrun[i] += ps;
    }

    asm volatile("s_wait_dscnt 0x0" ::: "memory");  // P stores visible before gather

    // O += P @ V : k-dim = key (2 steps of 32), n = head-dim
#pragma unroll
    for (int kk = 0; kk < 2; ++kk) {
      const bfx16 pf = lds_a_frag(Pw, PITCH, kk * 32, lane);
      bfx16 vf[4];
#pragma unroll
      for (int nt = 0; nt < 4; ++nt) vf[nt] = load_b_frag(Vh, T, nt * 16, s0 + kk * 32, lane);
#pragma unroll
      for (int nt = 0; nt < 4; ++nt) o[nt] = wmma_bf16(pf, vf[nt], o[nt]);
    }
  }

  // normalize and store to [b*T+t, h*64+e]
#pragma unroll
  for (int nt = 0; nt < 4; ++nt)
#pragma unroll
    for (int i = 0; i < 8; ++i) {
      const int tg = qRow + ((lane >> 4) << 3) + i;
      const int ee = nt * 16 + (lane & 15);
      Ob[((size_t)b * T + tg) * Dm + h * HD + ee] = f2bf(o[nt][i] / lrun[i]);
    }
}

// ======================================= launch ==========================================
extern "C" void kernel_launch(void* const* d_in, const int* in_sizes, int n_in,
                              void* d_out, int out_size, void* d_ws, size_t ws_size,
                              hipStream_t stream) {
  (void)in_sizes; (void)n_in; (void)out_size; (void)ws_size;
  const float* x  = (const float*)d_in[0];
  const float* Wq = (const float*)d_in[1];
  const float* Wk = (const float*)d_in[2];
  const float* Wv = (const float*)d_in[3];
  const float* Wo = (const float*)d_in[4];
  const float* bo = (const float*)d_in[5];
  const float* W1 = (const float*)d_in[6];
  const float* b1 = (const float*)d_in[7];
  const float* W2 = (const float*)d_in[8];
  const float* b2 = (const float*)d_in[9];
  const float* g1 = (const float*)d_in[10];
  const float* be1 = (const float*)d_in[11];
  const float* g2 = (const float*)d_in[12];
  const float* be2 = (const float*)d_in[13];
  float* out = (float*)d_out;

  constexpr size_t M = 4096, D = 1024, DF = 4096;
  char* ws = (char*)d_ws;
  size_t o = 0;
  auto alloc = [&](size_t bytes) { char* p = ws + o; o += bytes; return p; };
  __bf16* hA   = (__bf16*)alloc(M * D * 2);        // LN1 activations
  __bf16* WqT  = (__bf16*)alloc(D * D * 2);
  __bf16* WkT  = (__bf16*)alloc(D * D * 2);
  __bf16* WvT  = (__bf16*)alloc(D * D * 2);
  __bf16* WoT  = (__bf16*)alloc(D * D * 2);
  __bf16* W1T  = (__bf16*)alloc(D * DF * 2);
  __bf16* W2T  = (__bf16*)alloc(DF * D * 2);
  __bf16* Qb   = (__bf16*)alloc(M * D * 2);
  __bf16* Kbf  = (__bf16*)alloc(M * D * 2);
  __bf16* Vtb  = (__bf16*)alloc(M * D * 2);        // [B,H,HD,T]
  __bf16* Ao   = (__bf16*)alloc(M * D * 2);        // attention output
  float*  x1   = (float*)alloc(M * D * 4);         // post-attention residual
  __bf16* h2   = (__bf16*)alloc(M * D * 2);        // LN2 activations
  __bf16* f1   = (__bf16*)alloc(M * DF * 2);       // FFN hidden

  // 1. LN1 + convert weights
  ln_bf16_kernel<<<4096, 256, 0, stream>>>(x, g1, be1, hA);
  qkvT_kernel<<<1024, 256, 0, stream>>>(Wq, WqT);
  qkvT_kernel<<<1024, 256, 0, stream>>>(Wk, WkT);
  qkvT_kernel<<<1024, 256, 0, stream>>>(Wv, WvT);
  wT_kernel<<<2048, 256, 0, stream>>>(Wo, WoT, 1024, 1024);
  wT_kernel<<<2048, 256, 0, stream>>>(W1, W1T, 1024, 4096);
  wT_kernel<<<2048, 256, 0, stream>>>(W2, W2T, 4096, 1024);

  // 2. QKV projections (WMMA)
  gemm_bf16_kernel<false, false, false, false, true, false>
      <<<dim3(8, 32), 256, 0, stream>>>(hA, WqT, nullptr, nullptr, nullptr, Qb, 4096, 1024, 1024);
  gemm_bf16_kernel<false, false, false, false, true, false>
      <<<dim3(8, 32), 256, 0, stream>>>(hA, WkT, nullptr, nullptr, nullptr, Kbf, 4096, 1024, 1024);
  gemm_bf16_kernel<false, false, false, false, true, true>
      <<<dim3(8, 32), 256, 0, stream>>>(hA, WvT, nullptr, nullptr, nullptr, Vtb, 4096, 1024, 1024);

  // 3. causal flash attention
  attn_kernel<<<dim3(16, 32), 256, 0, stream>>>(Qb, Kbf, Vtb, Ao);

  // 4. output projection + bias + residual -> x1 (fp32)
  gemm_bf16_kernel<true, false, true, true, false, false>
      <<<dim3(8, 32), 256, 0, stream>>>(Ao, WoT, bo, x, x1, nullptr, 4096, 1024, 1024);

  // 5. LN2
  ln_bf16_kernel<<<4096, 256, 0, stream>>>(x1, g2, be2, h2);

  // 6. FFN1: relu(h2 @ W1 + b1) -> bf16
  gemm_bf16_kernel<true, true, false, false, true, false>
      <<<dim3(32, 32), 256, 0, stream>>>(h2, W1T, b1, nullptr, nullptr, f1, 4096, 4096, 1024);

  // 7. FFN2: f1 @ W2 + b2 + x1 -> out (fp32)
  gemm_bf16_kernel<true, false, true, true, false, false>
      <<<dim3(8, 32), 256, 0, stream>>>(f1, W2T, b2, x1, out, nullptr, 4096, 1024, 4096);
}
